// GetTopic_8830452761092
// MI455X (gfx1250) — matrix operand order repaired and checked by reading, not compiled, not run on hardware
//
#include <hip/hip_runtime.h>
#include <hip/hip_bf16.h>

typedef _Float16 h16;
typedef __attribute__((ext_vector_type(16))) _Float16 v16h;
typedef __attribute__((ext_vector_type(8)))  float    v8f;

#define BB 32
#define SS 512
#define WW 1024
#define HP 512
#define G4 2048
#define TOPICS 100
#define UU 32

__device__ __forceinline__ v8f wmma16(v16h a, v16h b, v8f c) {
  // D = A(16x32 f16) * B(32x16 f16) + C(16x16 f32)
  return __builtin_amdgcn_wmma_f32_16x16x32_f16(false, a, false, b, (short)0, c, false, false);
}
__device__ __forceinline__ float sigmoidf(float x) { return 1.f / (1.f + __expf(-x)); }

// ------------------------------------------------------------- bias combine
__global__ void bias_sum(const float* __restrict__ a, const float* __restrict__ b,
                         float* __restrict__ o, int n) {
  int i = blockIdx.x * blockDim.x + threadIdx.x;
  if (i < n) o[i] = a[i] + b[i];
}

// ---------------------------------------------- X -> A-fragment-major (f16)
// out frag idx = (mt*(WW/32)+kt)*32 + lane, 16 halves each.
// A 16x32 lane swizzle per ISA 7.12.2; GEMM row m: b = m&31, s = m>>5.
__global__ void swizzleA(const float* __restrict__ X, h16* __restrict__ out) {
  size_t i = (size_t)blockIdx.x * blockDim.x + threadIdx.x; // one per half
  int h    = (int)(i & 15);
  int lane = (int)((i >> 4) & 31);
  size_t t = i >> 9;
  int kt = (int)(t & 31);
  int mt = (int)(t >> 5);
  int ml = lane & 15, hl = lane >> 4;
  int v = h >> 1, odd = h & 1;
  int kk = (v < 4) ? (8 * hl + 2 * v) : (16 + 8 * hl + 2 * (v - 4));
  int k = kt * 32 + kk + odd;
  int m = mt * 16 + ml;
  out[i] = (h16)X[((size_t)(m & 31) * SS + (m >> 5)) * WW + k];
}

// ------------------------------------- W[N,K] -> B-fragment-major (f16)
// out frag idx = (nt*KT+kt)*32 + lane ; n = nt*16+(lane&15), k = kt*32+(h<8 ? 8*(lane>>4)+h : 16+8*(lane>>4)+h-8)
__global__ void swizzleB(const float* __restrict__ Wsrc, h16* __restrict__ out,
                         int KT, int K) {
  size_t i = (size_t)blockIdx.x * blockDim.x + threadIdx.x;
  int h    = (int)(i & 15);
  int lane = (int)((i >> 4) & 31);
  size_t t = i >> 9;
  int kt = (int)(t % KT);
  int nt = (int)(t / KT);
  int n = nt * 16 + (lane & 15);
  int k = kt * 32 + ((h < 8) ? (8 * (lane >> 4) + h) : (16 + 8 * (lane >> 4) + (h - 8)));
  out[i] = (h16)Wsrc[(size_t)n * K + k];
}

// -------------------------------------------------- input projection (GEMM)
// M=16384 (ts*32+b), N=2048, K=1024. All operands fragment-major.
// Output pre in C-fragment layout: idx = (((ts*2+mt)*4+g)*32+wr)*32+lane (v8f).
__global__ __launch_bounds__(256) void gemm_inproj(const h16* __restrict__ XA,
                                                   const h16* __restrict__ WB,
                                                   float* __restrict__ pre) {
  const int tid = threadIdx.x, lane = tid & 31, wv = tid >> 5;
  const int ts  = blockIdx.y;               // row block: rows ts*32..+31
  const int ntB = blockIdx.x * 32 + wv * 4; // 16-wide N-tile base
  const v16h* Af = (const v16h*)XA;
  const v16h* Bf = (const v16h*)WB;
  const int mt0 = ts * 2;

  v8f acc[2][4];
#pragma unroll
  for (int mt = 0; mt < 2; ++mt)
#pragma unroll
    for (int nf = 0; nf < 4; ++nf)
#pragma unroll
      for (int r = 0; r < 8; ++r) acc[mt][nf][r] = 0.f;

  for (int kt = 0; kt < WW / 32; ++kt) {
    v16h a0 = Af[((size_t)(mt0 + 0) * 32 + kt) * 32 + lane];
    v16h a1 = Af[((size_t)(mt0 + 1) * 32 + kt) * 32 + lane];
    if (kt + 2 < WW / 32)
      __builtin_prefetch(&Bf[((size_t)ntB * 32 + kt + 2) * 32 + lane], 0, 1);
#pragma unroll
    for (int nf = 0; nf < 4; ++nf) {
      v16h b = Bf[((size_t)(ntB + nf) * 32 + kt) * 32 + lane];
      acc[0][nf] = wmma16(a0, b, acc[0][nf]);
      acc[1][nf] = wmma16(a1, b, acc[1][nf]);
    }
  }

  v8f* po = (v8f*)pre;
#pragma unroll
  for (int mt = 0; mt < 2; ++mt)
#pragma unroll
    for (int nf = 0; nf < 4; ++nf) {
      int nt = ntB + nf;
      int g  = nt >> 5;   // gate
      int wr = nt & 31;   // wave-row within gate
      po[((((size_t)(ts * 2 + mt)) * 4 + g) * 32 + wr) * 32 + lane] = acc[mt][nf];
    }
}

// -------------------------------------------------------- recurrent LSTM
// One workgroup (32 waves) per direction; h kept in LDS in A-FRAGMENT layout
// so each step's A tiles are two clean ds_load_b128 per wave. Write-back of h
// applies the inverse A-swizzle. c-state lives in VGPRs for the whole seq.
__global__ __launch_bounds__(1024) void lstm_rec(const h16* __restrict__ WhhFB,
                                                 const h16* __restrict__ WhhBB,
                                                 const float* __restrict__ preF,
                                                 const float* __restrict__ preB,
                                                 const float* __restrict__ biasF,
                                                 const float* __restrict__ biasB,
                                                 float* __restrict__ hidden) {
  const int dir = blockIdx.x;
  const h16*   WB   = dir ? WhhBB : WhhFB;
  const float* pre  = dir ? preB  : preF;
  const float* bias = dir ? biasB : biasF;

  __shared__ h16 hA[2 * 16 * 32 * 16];   // 32 KB, A-fragment layout (mt,kt,lane,h)

  const int tid = threadIdx.x, lane = tid & 31, wv = tid >> 5;
  const int ml = lane & 15, hl = lane >> 4;
  const int j  = wv * 16 + ml;           // hidden unit this lane produces

  for (int i = tid; i < 2 * 16 * 32 * 16; i += 1024) hA[i] = (h16)0.f;

  // inverse A-swizzle constants for writing h[.., j] back into hA
  const int ktW = j >> 5;
  const int kk  = j & 31;
  const int hlT = (kk >> 3) & 1;
  const int he  = (((kk >> 4) & 1) * 4 + ((kk & 7) >> 1)) * 2 + (kk & 1);

  float bv[4];
#pragma unroll
  for (int g = 0; g < 4; ++g) bv[g] = bias[g * HP + j];

  float cst[16];
#pragma unroll
  for (int i = 0; i < 16; ++i) cst[i] = 0.f;

  __syncthreads();

  const v16h* Afs = (const v16h*)hA;
  const v16h* Bf  = (const v16h*)WB;
  const v8f*  pf  = (const v8f*)pre;

  for (int t = 0; t < SS; ++t) {
    const int ts = dir ? (SS - 1 - t) : t;

    v8f acc[2][4];
#pragma unroll
    for (int mt = 0; mt < 2; ++mt)
#pragma unroll
      for (int g = 0; g < 4; ++g) {
        v8f c = pf[((((size_t)(ts * 2 + mt)) * 4 + g) * 32 + wv) * 32 + lane];
#pragma unroll
        for (int r = 0; r < 8; ++r) c[r] += bv[g];
        acc[mt][g] = c;
      }

    for (int kt = 0; kt < HP / 32; ++kt) {
      v16h a0 = Afs[(0 * 16 + kt) * 32 + lane];
      v16h a1 = Afs[(1 * 16 + kt) * 32 + lane];
#pragma unroll
      for (int g = 0; g < 4; ++g) {
        v16h b = Bf[((size_t)(g * 32 + wv) * 16 + kt) * 32 + lane];
        acc[0][g] = wmma16(a0, b, acc[0][g]);
        acc[1][g] = wmma16(a1, b, acc[1][g]);
      }
    }
    __syncthreads();   // all waves done reading hA for this step

#pragma unroll
    for (int mt = 0; mt < 2; ++mt)
#pragma unroll
      for (int r = 0; r < 8; ++r) {
        float ig = sigmoidf(acc[mt][0][r]);
        float fg = sigmoidf(acc[mt][1][r]);
        float gg = tanhf(acc[mt][2][r]);
        float og = sigmoidf(acc[mt][3][r]);
        float c  = fg * cst[mt * 8 + r] + ig * gg;
        cst[mt * 8 + r] = c;
        float h  = og * tanhf(c);
        int m = mt * 16 + r + 8 * hl;             // batch index
        int laneT = hlT * 16 + (m & 15);
        hA[((mt * 16 + ktW) * 32 + laneT) * 16 + he] = (h16)h;
        hidden[((size_t)m * SS + ts) * WW + dir * HP + j] = h;
      }
    __syncthreads();   // new h visible before next step's WMMA reads
  }
}

// ------------------------------------------------ segment max pool (clamped 0)
__global__ void pool_kernel(const float* __restrict__ hidden,
                            const int* __restrict__ uid,
                            float* __restrict__ mp) {
  const int u = blockIdx.x + 1;  // 1..32
  const int b = blockIdx.y;
  __shared__ int su[SS];
  for (int i = threadIdx.x; i < SS; i += blockDim.x) su[i] = uid[b * SS + i];
  __syncthreads();
  for (int w = threadIdx.x; w < WW; w += blockDim.x) {
    float mx = 0.f;  // clamp-at-zero == jnp.maximum(mp, 0) and empty segments
    for (int s = 0; s < SS; ++s)
      if (su[s] == u) mx = fmaxf(mx, hidden[((size_t)b * SS + s) * WW + w]);
    mp[((size_t)b * UU + (u - 1)) * WW + w] = mx;
  }
}

// ----------------------------------------- topic logits -> softmax -> embed
__global__ void topic_kernel(const float* __restrict__ mp,
                             const float* __restrict__ tw,
                             const float* __restrict__ tb,
                             const float* __restrict__ tt,
                             float* __restrict__ emb) {
  const int row = blockIdx.x;                 // b*32 + u
  const float* m = mp + (size_t)row * WW;
  __shared__ float prob[128];
  __shared__ float inv_s;
  const int tid = threadIdx.x;
  if (tid < 128) {
    float logit = -1e30f;
    if (tid < TOPICS) {
      const float* w = tw + (size_t)tid * WW;
      float s = 0.f;
      for (int k = 0; k < WW; ++k) s += m[k] * w[k];
      logit = s + tb[tid];
    }
    prob[tid] = logit;
  }
  __syncthreads();
  if (tid == 0) {
    float mx = -1e30f;
    for (int i = 0; i < TOPICS; ++i) mx = fmaxf(mx, prob[i]);
    float sm = 0.f;
    for (int i = 0; i < TOPICS; ++i) { float e = __expf(prob[i] - mx); prob[i] = e; sm += e; }
    inv_s = 1.f / sm;
  }
  __syncthreads();
  const float inv = inv_s;
  for (int w = tid; w < WW; w += blockDim.x) {
    float s = 0.f;
    for (int t = 0; t < TOPICS; ++t) s += prob[t] * tt[(size_t)t * WW + w];
    emb[(size_t)row * WW + w] = s * inv;
  }
}

// ------------------------------------------------------- weighted gather out
__global__ void gather_kernel(const float* __restrict__ emb,
                              const int* __restrict__ uid,
                              float* __restrict__ out) {
  const int tok = blockIdx.x;        // b*512 + s
  const int b   = tok >> 9;
  const int id  = uid[tok];
  const float wgt = (id > 0) ? 1.f : (id < 0 ? 2.f : 0.f);
  int idx = id < 0 ? -id : id;
  idx -= 1;
  if (idx < 0) idx = 0;
  if (idx > UU - 1) idx = UU - 1;
  const float* e = emb + ((size_t)b * UU + idx) * WW;
  float* o = out + (size_t)tok * WW;
  for (int w = threadIdx.x; w < WW; w += blockDim.x) o[w] = e[w] * wgt;
}

// --------------------------------------------------------------- launcher
extern "C" void kernel_launch(void* const* d_in, const int* in_sizes, int n_in,
                              void* d_out, int out_size, void* d_ws, size_t ws_size,
                              hipStream_t stream) {
  (void)in_sizes; (void)n_in; (void)out_size; (void)ws_size;
  const float* X    = (const float*)d_in[0];
  const int*   uid  = (const int*)  d_in[1];
  const float* WihF = (const float*)d_in[2];
  const float* WhhF = (const float*)d_in[3];
  const float* bihF = (const float*)d_in[4];
  const float* bhhF = (const float*)d_in[5];
  const float* WihB = (const float*)d_in[6];
  const float* WhhB = (const float*)d_in[7];
  const float* bihB = (const float*)d_in[8];
  const float* bhhB = (const float*)d_in[9];
  const float* tw   = (const float*)d_in[10];
  const float* tb   = (const float*)d_in[11];
  const float* tt   = (const float*)d_in[12];
  float* out = (float*)d_out;

  char* p = (char*)d_ws;
  auto take = [&](size_t bytes) { char* r = p; p += (bytes + 255) & ~(size_t)255; return r; };
  h16*   XA    = (h16*)  take((size_t)BB * SS * WW * 2);   // A-frag layout
  h16*   WihFh = (h16*)  take((size_t)G4 * WW * 2);        // B-frag layout
  h16*   WihBh = (h16*)  take((size_t)G4 * WW * 2);
  h16*   WhhFh = (h16*)  take((size_t)G4 * HP * 2);
  h16*   WhhBh = (h16*)  take((size_t)G4 * HP * 2);
  float* bF    = (float*)take((size_t)G4 * 4);
  float* bB    = (float*)take((size_t)G4 * 4);
  float* preF  = (float*)take((size_t)BB * SS * G4 * 4);   // C-frag layout
  float* preB  = (float*)take((size_t)BB * SS * G4 * 4);
  float* hid   = (float*)take((size_t)BB * SS * WW * 4);
  float* mp    = (float*)take((size_t)BB * UU * WW * 4);
  float* emb   = (float*)take((size_t)BB * UU * WW * 4);

  auto cdiv = [](size_t a, size_t b) { return (unsigned)((a + b - 1) / b); };

  const size_t nX = (size_t)BB * SS * WW;
  swizzleA<<<cdiv(nX, 256), 256, 0, stream>>>(X, XA);
  swizzleB<<<cdiv((size_t)G4 * WW, 256), 256, 0, stream>>>(WihF, WihFh, WW / 32, WW);
  swizzleB<<<cdiv((size_t)G4 * WW, 256), 256, 0, stream>>>(WihB, WihBh, WW / 32, WW);
  swizzleB<<<cdiv((size_t)G4 * HP, 256), 256, 0, stream>>>(WhhF, WhhFh, HP / 32, HP);
  swizzleB<<<cdiv((size_t)G4 * HP, 256), 256, 0, stream>>>(WhhB, WhhBh, HP / 32, HP);
  bias_sum<<<G4 / 256, 256, 0, stream>>>(bihF, bhhF, bF, G4);
  bias_sum<<<G4 / 256, 256, 0, stream>>>(bihB, bhhB, bB, G4);

  dim3 gg(G4 / 512, (BB * SS) / 32);  // (4, 512)
  gemm_inproj<<<gg, 256, 0, stream>>>(XA, WihFh, preF);
  gemm_inproj<<<gg, 256, 0, stream>>>(XA, WihBh, preB);

  lstm_rec<<<2, 1024, 0, stream>>>(WhhFh, WhhBh, preF, preB, bF, bB, hid);

  pool_kernel<<<dim3(UU, BB), 256, 0, stream>>>(hid, uid, mp);
  topic_kernel<<<BB * UU, 256, 0, stream>>>(mp, tw, tb, tt, emb);
  gather_kernel<<<BB * SS, 256, 0, stream>>>(emb, uid, out);
}
